// RSNN_eprop_88845693485082
// MI455X (gfx1250) — compile-verified, matrix operand order checked by reading, not run on hardware
//
#include <hip/hip_runtime.h>

// ---------------------------------------------------------------------------
// LIF RNN forward for MI455X (gfx1250, wave32, WMMA 16x16x32 bf16).
//
//  1) Convert x / W_in / W_rec(zero-diag) / W_out(pad to 32 rows) to bf16.
//  2) One parallel bf16 WMMA GEMM: in_cur[t,b,j] = x[t] @ W_in^T (32000x704x1024)
//  3) ONE persistent kernel (33 blocks, grid spin-barrier) runs all 500 steps:
//       - blocks 0..31: z@W_rec^T (WMMA, B staged once in LDS) + LIF epilogue,
//         membrane potential v kept in VGPRs for the whole sequence.
//       - block 32:     vo = kappa*vo + z@W_out^T (vo in VGPRs), writes out[t].
//       - spikes z ping-pong through global (the only per-step shared state).
//  4) In-place softmax over the 20 outputs.
// ---------------------------------------------------------------------------

typedef __attribute__((ext_vector_type(16))) __bf16 v16bf;
typedef __attribute__((ext_vector_type(8)))  __bf16 v8bf;
typedef __attribute__((ext_vector_type(8)))  float  v8f;

#define LIF_THR   0.6f
#define LIF_ALPHA 0.9512294245007140f   // exp(-1/20)
#define LIF_KAPPA 0.9512294245007140f   // exp(-1/20)

constexpr int N_T       = 500;
constexpr int N_B       = 64;
constexpr int N_IN      = 700;
constexpr int N_REC     = 1024;
constexpr int N_OUT     = 20;
constexpr int K_IN_PAD  = 704;   // 700 padded to multiple of 32
constexpr int N_OUT_PAD = 32;    // 20 padded to multiple of 16
constexpr int LDS_LD    = 1032;  // 1024 + 8 bf16 pad -> row stride 516 dwords
                                 // (516 mod 64 = 4 -> conflict-free 16-lane gather)
constexpr int NBLOCKS   = 33;    // 32 recurrent + 1 output

// --- WMMA helpers -----------------------------------------------------------

// 16x32 bf16 fragment (A layout; same layout serves B when "B" is W (N x K)
// row-major, i.e. W^T column-major).
// Lane L<16  holds row (row0+L),    K = kbase + {0..7, 16..23}
// Lane L>=16 holds row (row0+L-16), K = kbase + {8..15, 24..31}
template <typename PtrT>
__device__ inline v16bf load_frag(PtrT base, int row0, int ld, int kbase, int lane) {
  const int half = lane >> 4;
  const int r    = lane & 15;
  auto p = base + (size_t)(row0 + r) * ld + kbase + half * 8;
  union { v16bf f; v8bf h[2]; } u;
  u.h[0] = *reinterpret_cast<const v8bf*>(&p[0]);     // b128 load
  u.h[1] = *reinterpret_cast<const v8bf*>(&p[16]);    // b128 load
  return u.f;
}

__device__ inline v8f wmma_bf16(v16bf a, v16bf b, v8f c) {
  return __builtin_amdgcn_wmma_f32_16x16x32_bf16(
      /*neg_a=*/false, a, /*neg_b=*/false, b,
      /*c_mod=*/(short)0, c, /*reuse_a=*/false, /*reuse_b=*/false);
}

// --- conversion / init kernels ---------------------------------------------

__global__ void conv_x_kernel(const float* __restrict__ x, __bf16* __restrict__ xp) {
  size_t i = (size_t)blockIdx.x * blockDim.x + threadIdx.x;
  const size_t total = (size_t)N_T * N_B * K_IN_PAD;
  if (i >= total) return;
  int k = (int)(i % K_IN_PAD);
  size_t m = i / K_IN_PAD;
  float v = (k < N_IN) ? x[m * N_IN + k] : 0.0f;
  xp[i] = (__bf16)v;
}

__global__ void conv_win_kernel(const float* __restrict__ w, __bf16* __restrict__ wp) {
  size_t i = (size_t)blockIdx.x * blockDim.x + threadIdx.x;
  const size_t total = (size_t)N_REC * K_IN_PAD;
  if (i >= total) return;
  int k = (int)(i % K_IN_PAD);
  size_t n = i / K_IN_PAD;
  float v = (k < N_IN) ? w[n * N_IN + k] : 0.0f;
  wp[i] = (__bf16)v;
}

__global__ void conv_wrec_kernel(const float* __restrict__ w, __bf16* __restrict__ wp) {
  size_t i = (size_t)blockIdx.x * blockDim.x + threadIdx.x;
  const size_t total = (size_t)N_REC * N_REC;
  if (i >= total) return;
  int k = (int)(i % N_REC);
  int n = (int)(i / N_REC);
  float v = (n == k) ? 0.0f : w[i];      // zero self-connections
  wp[i] = (__bf16)v;
}

__global__ void conv_wout_kernel(const float* __restrict__ w, __bf16* __restrict__ wp) {
  size_t i = (size_t)blockIdx.x * blockDim.x + threadIdx.x;
  const size_t total = (size_t)N_OUT_PAD * N_REC;
  if (i >= total) return;
  int k = (int)(i % N_REC);
  int n = (int)(i / N_REC);
  float v = (n < N_OUT) ? w[(size_t)n * N_REC + k] : 0.0f;
  wp[i] = (__bf16)v;
}

__global__ void init_state_kernel(__bf16* __restrict__ z0, __bf16* __restrict__ z1,
                                  unsigned* __restrict__ bar) {
  int i = blockIdx.x * blockDim.x + threadIdx.x;
  if (i < N_B * N_REC) {
    z0[i] = (__bf16)0.0f;
    z1[i] = (__bf16)0.0f;
  }
  if (i == 0) *bar = 0u;
}

// --- phase 2: parallel input projection  in_cur = X @ W_in^T ----------------
// grid = (500, 32), block = 256 (8 waves); each wave: one 16x16 tile, K=704.
__global__ void gemm_incur_kernel(const __bf16* __restrict__ X,
                                  const __bf16* __restrict__ Win,
                                  float* __restrict__ C) {
  const int wave = threadIdx.x >> 5;
  const int lane = threadIdx.x & 31;
  const int mbase = blockIdx.x * 64 + (wave & 3) * 16;    // 32000 rows
  const int nbase = blockIdx.y * 32 + (wave >> 2) * 16;   // 1024 cols
  v8f c = {};
#pragma unroll 2
  for (int k = 0; k < K_IN_PAD; k += 32) {
    v16bf a = load_frag(X,   mbase, K_IN_PAD, k, lane);
    v16bf b = load_frag(Win, nbase, K_IN_PAD, k, lane);
    c = wmma_bf16(a, b, c);
  }
  const int col    = nbase + (lane & 15);
  const int rowoff = mbase + ((lane >> 4) << 3);
#pragma unroll
  for (int r = 0; r < 8; ++r)
    C[(size_t)(rowoff + r) * N_REC + col] = c[r];
}

// --- phase 3: persistent kernel, all 500 LIF steps --------------------------
// grid = 33 blocks x 256 threads; all blocks resident -> spin barrier is safe.
__global__ void lif_persistent_kernel(const float* __restrict__ in_cur,
                                      const __bf16* __restrict__ Wrec,
                                      const __bf16* __restrict__ Wout,
                                      __bf16* __restrict__ zb0,
                                      __bf16* __restrict__ zb1,
                                      float* __restrict__ out,
                                      unsigned* __restrict__ bar) {
  extern __shared__ __bf16 lds_w[];     // [32][LDS_LD] = 66,048 B

  const int wave = threadIdx.x >> 5;
  const int lane = threadIdx.x & 31;
  const int r16  = lane & 15;
  const int half = lane >> 4;
  const bool is_out_block = (blockIdx.x == 32);

  // ---- stage this block's 32 weight rows into LDS (once, reused 500x) ----
  {
    const __bf16* src = is_out_block ? Wout
                                     : (Wrec + (size_t)blockIdx.x * 32 * N_REC);
    for (int idx = threadIdx.x; idx < 32 * (N_REC / 8); idx += blockDim.x) {
      const int row   = idx >> 7;          // /128 chunks of 8 bf16 per row
      const int chunk = idx & 127;
      *reinterpret_cast<v8bf*>(&lds_w[row * LDS_LD + chunk * 8]) =
          *reinterpret_cast<const v8bf*>(&src[(size_t)row * N_REC + chunk * 8]);
    }
  }
  __syncthreads();

  // ---- per-thread tile ownership (fixed for the whole sequence) ----
  const int mbase  = (wave & 3) * 16;                       // batch rows
  const int nloc   = (wave >> 2) * 16;                      // 0 or 16 (LDS-local)
  const int nbase  = is_out_block ? nloc : blockIdx.x * 32 + nloc;
  const int col    = nbase + r16;
  const int rowoff = mbase + half * 8;

  // persistent register state
  v8f vstate  = {};   // membrane potential (recurrent blocks)
  v8f zstate  = {};   // this thread's own spikes (0/1), for the -thr*z term
  v8f vostate = {};   // output trace (block 32)

  for (int t = 0; t < N_T; ++t) {
    const __bf16* zcur  = (t & 1) ? zb1 : zb0;
    __bf16*       znext = (t & 1) ? zb0 : zb1;

    if (!is_out_block) {
      if (t < N_T - 1) {                                    // z_500 not needed
        v8f c = {};
#pragma unroll 2
        for (int k = 0; k < N_REC; k += 32) {
          v16bf a = load_frag(zcur,  mbase, N_REC,  k, lane);   // global
          v16bf b = load_frag(lds_w, nloc,  LDS_LD, k, lane);   // LDS
          c = wmma_bf16(a, b, c);
        }
        const float* inc = in_cur + (size_t)t * N_B * N_REC;
#pragma unroll
        for (int r = 0; r < 8; ++r) {
          const size_t idx = (size_t)(rowoff + r) * N_REC + col;
          const float vn = LIF_ALPHA * vstate[r] + c[r] + inc[idx]
                           - LIF_THR * zstate[r];
          const float zn = (vn > LIF_THR) ? 1.0f : 0.0f;
          vstate[r] = vn;
          zstate[r] = zn;
          znext[idx] = (__bf16)zn;
        }
      }
    } else {
      // vo_t = kappa * vo_{t-1} + z_t @ Wout^T  (z_0 == 0 -> out[0] == 0)
      v8f c = {};
#pragma unroll 2
      for (int k = 0; k < N_REC; k += 32) {
        v16bf a = load_frag(zcur,  mbase, N_REC,  k, lane);     // global
        v16bf b = load_frag(lds_w, nloc,  LDS_LD, k, lane);     // LDS
        c = wmma_bf16(a, b, c);
      }
#pragma unroll
      for (int r = 0; r < 8; ++r)
        vostate[r] = LIF_KAPPA * vostate[r] + c[r];
      if (col < N_OUT) {                                    // post-WMMA divergence
        float* o = out + (size_t)t * N_B * N_OUT;
#pragma unroll
        for (int r = 0; r < 8; ++r)
          o[(rowoff + r) * N_OUT + col] = vostate[r];
      }
    }

    // ---- device-wide barrier (monotonic counter, release/acquire) ----
    __threadfence();
    __syncthreads();
    if (threadIdx.x == 0) {
      __hip_atomic_fetch_add(bar, 1u, __ATOMIC_RELEASE, __HIP_MEMORY_SCOPE_AGENT);
      const unsigned target = (unsigned)(t + 1) * NBLOCKS;
      while (__hip_atomic_load(bar, __ATOMIC_ACQUIRE, __HIP_MEMORY_SCOPE_AGENT)
             < target) {
        __builtin_amdgcn_s_sleep(2);
      }
    }
    __syncthreads();
  }
}

// --- phase 4: softmax over the 20 outputs, in place -------------------------
__global__ void softmax_kernel(float* __restrict__ out) {
  int row = blockIdx.x * blockDim.x + threadIdx.x;
  if (row >= N_T * N_B) return;
  float* p = out + (size_t)row * N_OUT;
  float m = p[0];
#pragma unroll
  for (int i = 1; i < N_OUT; ++i) m = fmaxf(m, p[i]);
  float e[N_OUT];
  float s = 0.0f;
#pragma unroll
  for (int i = 0; i < N_OUT; ++i) { e[i] = __expf(p[i] - m); s += e[i]; }
  const float inv = 1.0f / s;
#pragma unroll
  for (int i = 0; i < N_OUT; ++i) p[i] = e[i] * inv;
}

// ---------------------------------------------------------------------------

extern "C" void kernel_launch(void* const* d_in, const int* in_sizes, int n_in,
                              void* d_out, int out_size, void* d_ws, size_t ws_size,
                              hipStream_t stream) {
  const float* x     = (const float*)d_in[0];   // (500, 64, 700)
  const float* w_in  = (const float*)d_in[1];   // (1024, 700)
  const float* w_rec = (const float*)d_in[2];   // (1024, 1024)
  const float* w_out = (const float*)d_in[3];   // (20, 1024)
  float* out = (float*)d_out;                   // (500, 64, 20)

  // workspace layout
  char* ws = (char*)d_ws;
  auto alignup = [](size_t v) { return (v + 255) & ~(size_t)255; };
  size_t off = 0;
  float*  in_cur = (float*) (ws + off); off = alignup(off + (size_t)N_T * N_B * N_REC * 4);
  __bf16* xp     = (__bf16*)(ws + off); off = alignup(off + (size_t)N_T * N_B * K_IN_PAD * 2);
  __bf16* winp   = (__bf16*)(ws + off); off = alignup(off + (size_t)N_REC * K_IN_PAD * 2);
  __bf16* wrecp  = (__bf16*)(ws + off); off = alignup(off + (size_t)N_REC * N_REC * 2);
  __bf16* woutp  = (__bf16*)(ws + off); off = alignup(off + (size_t)N_OUT_PAD * N_REC * 2);
  __bf16* zbuf0  = (__bf16*)(ws + off); off = alignup(off + (size_t)N_B * N_REC * 2);
  __bf16* zbuf1  = (__bf16*)(ws + off); off = alignup(off + (size_t)N_B * N_REC * 2);
  unsigned* bar  = (unsigned*)(ws + off); off = alignup(off + 256);

  // 1) conversions + state init
  {
    const size_t tx = (size_t)N_T * N_B * K_IN_PAD;
    conv_x_kernel<<<(unsigned)((tx + 255) / 256), 256, 0, stream>>>(x, xp);
    const size_t ti = (size_t)N_REC * K_IN_PAD;
    conv_win_kernel<<<(unsigned)((ti + 255) / 256), 256, 0, stream>>>(w_in, winp);
    const size_t tr = (size_t)N_REC * N_REC;
    conv_wrec_kernel<<<(unsigned)((tr + 255) / 256), 256, 0, stream>>>(w_rec, wrecp);
    const size_t to = (size_t)N_OUT_PAD * N_REC;
    conv_wout_kernel<<<(unsigned)((to + 255) / 256), 256, 0, stream>>>(w_out, woutp);
    init_state_kernel<<<(N_B * N_REC + 255) / 256, 256, 0, stream>>>(zbuf0, zbuf1, bar);
  }

  // 2) parallel input projection (bf16 WMMA GEMM, 32000 x 704 x 1024)
  gemm_incur_kernel<<<dim3(N_T, N_REC / 32), 256, 0, stream>>>(xp, winp, in_cur);

  // 3) ONE persistent launch for all 500 sequential LIF steps
  const size_t lds_bytes = (size_t)32 * LDS_LD * sizeof(__bf16);  // 66,048 B
  lif_persistent_kernel<<<NBLOCKS, 256, lds_bytes, stream>>>(
      in_cur, wrecp, woutp, zbuf0, zbuf1, out, bar);

  // 4) softmax over the 20 output units, in place
  softmax_kernel<<<(N_T * N_B + 255) / 256, 256, 0, stream>>>(out);
}
